// TriFeatureGSLGNN_5368709120675
// MI455X (gfx1250) — matrix operand order repaired and checked by reading, not compiled, not run on hardware
//
#include <hip/hip_runtime.h>
#include <hip/hip_bf16.h>

#define NN 8192
#define DP 128
#define HH 256
#define CC 16
#define COMB 768
#define LBL 64
#define BN_EPS 1e-5f
#define THRESH 0.1f

typedef __attribute__((ext_vector_type(16))) __bf16 v16bf;
typedef __attribute__((ext_vector_type(8)))  float  v8f;

// ---------------------------------------------------------------------------
// Build A_hat (bf16), binary A_gcn with self loops (bf16), and dinv = deg^-1/2
// ---------------------------------------------------------------------------
__global__ __launch_bounds__(256)
void build_adj_kernel(const float* __restrict__ L,
                      __hip_bfloat16* __restrict__ Ahat,
                      __hip_bfloat16* __restrict__ Agcn,
                      float* __restrict__ dinv)
{
    int i = blockIdx.x;
    float deg = 0.f;
    for (int j = threadIdx.x; j < NN; j += blockDim.x) {
        float l = 0.5f * (L[(size_t)i * NN + j] + L[(size_t)j * NN + i]);
        float a = 1.f / (1.f + __expf(-l));
        Ahat[(size_t)i * NN + j] = __float2bfloat16(a);
        float g = (i == j) ? 1.f : ((a > THRESH) ? 1.f : 0.f);
        Agcn[(size_t)i * NN + j] = __float2bfloat16(g);
        deg += g;
    }
    __shared__ float red[256];
    red[threadIdx.x] = deg;
    __syncthreads();
    for (int s = blockDim.x >> 1; s > 0; s >>= 1) {
        if (threadIdx.x < s) red[threadIdx.x] += red[threadIdx.x + s];
        __syncthreads();
    }
    if (threadIdx.x == 0) dinv[i] = rsqrtf(red[0]);
}

// ---------------------------------------------------------------------------
// Repack f32 [K,Ncols] (optionally transposed / row-scaled) into this file's (contiguous-k) wave32
// WMMA B-operand layout: dword idx = ((ct*KC + kc)*32 + lane)*8 + v, where
// lanes 0-15 hold K=2v,2v+1 and lanes 16-31 hold K=16+2v,17+2v of the chunk.
// ---------------------------------------------------------------------------
__global__ __launch_bounds__(256)
void pack_b_kernel(const float* __restrict__ src, int K, int Ncols, int ld,
                   int col0, int transposed, const float* __restrict__ rowscale,
                   __hip_bfloat16* __restrict__ dst)
{
    int idx = blockIdx.x * blockDim.x + threadIdx.x;
    int KC = K >> 5;
    int total = (Ncols >> 4) * KC * 32 * 8;
    if (idx >= total) return;
    int v    = idx & 7;
    int lane = (idx >> 3) & 31;
    int rest = idx >> 8;
    int kc = rest % KC;
    int ct = rest / KC;
    int n = (lane & 15) + (ct << 4) + col0;
    int k = (kc << 5) + ((lane & 16) ? 16 : 0) + (v << 1);
    float s0, s1;
    if (transposed) {
        s0 = src[(size_t)n * ld + k];
        s1 = src[(size_t)n * ld + k + 1];
    } else {
        s0 = src[(size_t)k * ld + n];
        s1 = src[(size_t)(k + 1) * ld + n];
    }
    if (rowscale) { s0 *= rowscale[k]; s1 *= rowscale[k + 1]; }
    dst[(size_t)idx * 2]     = __float2bfloat16(s0);
    dst[(size_t)idx * 2 + 1] = __float2bfloat16(s1);
}

// ---------------------------------------------------------------------------
// Single-tile WMMA K-loop (used by the diffusion kernel, Ncols == 16).
// ---------------------------------------------------------------------------
__device__ __forceinline__
v8f wmma_tile(const __hip_bfloat16* __restrict__ A, int lda,
              const __hip_bfloat16* __restrict__ Bpack,
              int tileM, int ct, int K, int lane)
{
    v8f acc = {};
    int rowA  = tileM * 16 + (lane & 15);
    int aoff0 = (lane & 16) ? 16 : 0;    // lanes 16-31: K offset +16 / +24
    int KC = K >> 5;
    const float4* Arow = reinterpret_cast<const float4*>(A + (size_t)rowA * lda);
    const float4* Bp   = reinterpret_cast<const float4*>(
        Bpack + ((size_t)ct * KC * 32) * 16);
    for (int kc = 0; kc < KC; ++kc) {
        int k0 = kc << 5;
        union { v16bf v; float4 f[2]; } a;
        a.f[0] = Arow[(k0 + aoff0) >> 3];
        a.f[1] = Arow[(k0 + aoff0 + 8) >> 3];
        union { v16bf v; float4 f[2]; } b;
        const float4* bl = Bp + ((size_t)kc * 32 + lane) * 2;
        b.f[0] = bl[0];
        b.f[1] = bl[1];
        __builtin_prefetch((const void*)(Arow + ((k0) >> 3) + 64), 0, 1);
        acc = __builtin_amdgcn_wmma_f32_16x16x32_bf16(
            false, a.v, false, b.v, (short)0, acc, false, false);
    }
    return acc;
}

// ---------------------------------------------------------------------------
// One damped-diffusion hop with fused row softmax (C == 16 == one WMMA tile).
// out column slice [hop*16 .. hop*16+15] of label_comb [N, 64].
// ---------------------------------------------------------------------------
__global__ __launch_bounds__(256)
void diffusion_kernel(const __hip_bfloat16* __restrict__ Ahat,
                      const __hip_bfloat16* __restrict__ Bpack,
                      const float* __restrict__ Y,
                      const float* __restrict__ alpha_p,
                      float* __restrict__ outLbl, int hop)
{
    int wid  = blockIdx.x * (blockDim.x >> 5) + (threadIdx.x >> 5);
    int lane = threadIdx.x & 31;
    if (wid >= NN / 16) return;
    v8f acc = wmma_tile(Ahat, NN, Bpack, wid, 0, NN, lane);
    float alpha = 1.f / (1.f + __expf(-alpha_p[0]));
    int n = lane & 15;
    int rbase = (lane & 16) ? 8 : 0;
#pragma unroll
    for (int r = 0; r < 8; ++r) {
        int row = wid * 16 + rbase + r;
        float d = alpha * acc[r] + (1.f - alpha) * Y[(size_t)row * CC + n];
        float m = d;
#pragma unroll
        for (int msk = 1; msk < 16; msk <<= 1)
            m = fmaxf(m, __shfl_xor(m, msk, 32));
        float e = __expf(d - m);
        float s = e;
#pragma unroll
        for (int msk = 1; msk < 16; msk <<= 1)
            s += __shfl_xor(s, msk, 32);
        outLbl[(size_t)row * LBL + hop * CC + n] = e / s;
    }
}

// ---------------------------------------------------------------------------
// Generic bf16 WMMA GEMM: each wave computes NT adjacent 16x16 column tiles,
// reusing one A fragment across NT WMMAs (NT=4 for H=256 outputs -> 4x less
// A traffic; NT=1 for C=16 outputs). rowscale / bias / relu epilogue, f32
// and/or bf16 output.
// ---------------------------------------------------------------------------
template <int NT>
__global__ __launch_bounds__(256)
void gemm_bf16_kernel(const __hip_bfloat16* __restrict__ A, int lda,
                      const __hip_bfloat16* __restrict__ Bpack,
                      int Mtiles, int K, int Ntiles,
                      const float* __restrict__ rowscale,
                      const float* __restrict__ bias, int relu,
                      float* __restrict__ outF,
                      __hip_bfloat16* __restrict__ outB, int ldo)
{
    int wid  = blockIdx.x * (blockDim.x >> 5) + (threadIdx.x >> 5);
    int lane = threadIdx.x & 31;
    int groups = Ntiles / NT;
    if (wid >= Mtiles * groups) return;
    int tileM = wid % Mtiles;
    int ctg   = wid / Mtiles;
    int KC    = K >> 5;

    v8f acc[NT];
#pragma unroll
    for (int t = 0; t < NT; ++t) acc[t] = (v8f){};

    int rowA  = tileM * 16 + (lane & 15);
    int aoff0 = (lane & 16) ? 16 : 0;
    const float4* Arow = reinterpret_cast<const float4*>(A + (size_t)rowA * lda);
    const float4* Bp   = reinterpret_cast<const float4*>(Bpack);

    for (int kc = 0; kc < KC; ++kc) {
        int k0 = kc << 5;
        union { v16bf v; float4 f[2]; } a;
        a.f[0] = Arow[(k0 + aoff0) >> 3];
        a.f[1] = Arow[(k0 + aoff0 + 8) >> 3];
        __builtin_prefetch((const void*)(Arow + (k0 >> 3) + 64), 0, 1);
#pragma unroll
        for (int t = 0; t < NT; ++t) {
            int ct = ctg * NT + t;
            union { v16bf v; float4 f[2]; } b;
            const float4* bl = Bp + ((size_t)(ct * KC + kc) * 32 + lane) * 2;
            b.f[0] = bl[0];
            b.f[1] = bl[1];
            acc[t] = __builtin_amdgcn_wmma_f32_16x16x32_bf16(
                false, a.v, false, b.v, (short)0, acc[t], false, false);
        }
    }

    int rbase = (lane & 16) ? 8 : 0;
#pragma unroll
    for (int t = 0; t < NT; ++t) {
        int n = (lane & 15) + ((ctg * NT + t) << 4);
#pragma unroll
        for (int r = 0; r < 8; ++r) {
            int row = tileM * 16 + rbase + r;
            float v = acc[t][r];
            if (rowscale) v *= rowscale[row];
            if (bias)     v += bias[n];
            if (relu)     v = fmaxf(v, 0.f);
            if (outF) outF[(size_t)row * ldo + n] = v;
            if (outB) outB[(size_t)row * ldo + n] = __float2bfloat16(v);
        }
    }
}

// ---------------------------------------------------------------------------
// comb = concat(relu(Xp@Wp^T+bp), relu(Xs@Ws^T+bs), relu(lbl@Wl^T+bl)) -> bf16
// ---------------------------------------------------------------------------
__global__ __launch_bounds__(256)
void comb_kernel(const float* __restrict__ Xp, const float* __restrict__ Xs,
                 const float* __restrict__ lbl,
                 const float* __restrict__ Wp, const float* __restrict__ bp,
                 const float* __restrict__ Ws, const float* __restrict__ bs,
                 const float* __restrict__ Wl, const float* __restrict__ bl,
                 __hip_bfloat16* __restrict__ comb)
{
    int idx = blockIdx.x * blockDim.x + threadIdx.x;
    if (idx >= NN * COMB) return;
    int i = idx / COMB, c = idx % COMB;
    float v;
    if (c < HH) {
        const float* w = Wp + (size_t)c * DP;
        const float* x = Xp + (size_t)i * DP;
        float s = bp[c];
        for (int k = 0; k < DP; ++k) s += x[k] * w[k];
        v = s;
    } else if (c < 2 * HH) {
        int h = c - HH;
        v = Xs[i] * Ws[h] + bs[h];
    } else {
        int h = c - 2 * HH;
        const float* w = Wl + (size_t)h * LBL;
        const float* x = lbl + (size_t)i * LBL;
        float s = bl[h];
        for (int k = 0; k < LBL; ++k) s += x[k] * w[k];
        v = s;
    }
    comb[idx] = __float2bfloat16(fmaxf(v, 0.f));
}

// ---------------------------------------------------------------------------
// Batch-norm statistics (biased var) over axis 0 of h [N, H]
// ---------------------------------------------------------------------------
__global__ __launch_bounds__(256)
void bn_stats_kernel(const float* __restrict__ h, float* __restrict__ mu,
                     float* __restrict__ var)
{
    int c = blockIdx.x;
    float s = 0.f, s2 = 0.f;
    for (int i = threadIdx.x; i < NN; i += blockDim.x) {
        float x = h[(size_t)i * HH + c];
        s += x; s2 += x * x;
    }
    __shared__ float r1[256], r2[256];
    r1[threadIdx.x] = s; r2[threadIdx.x] = s2;
    __syncthreads();
    for (int st = blockDim.x >> 1; st > 0; st >>= 1) {
        if (threadIdx.x < st) {
            r1[threadIdx.x] += r1[threadIdx.x + st];
            r2[threadIdx.x] += r2[threadIdx.x + st];
        }
        __syncthreads();
    }
    if (threadIdx.x == 0) {
        float m = r1[0] / (float)NN;
        mu[c]  = m;
        var[c] = r2[0] / (float)NN - m * m;
    }
}

__global__ __launch_bounds__(256)
void bn_apply_kernel(const float* __restrict__ h, const float* __restrict__ mu,
                     const float* __restrict__ var,
                     const float* __restrict__ gamma,
                     const float* __restrict__ beta,
                     __hip_bfloat16* __restrict__ out)
{
    int idx = blockIdx.x * blockDim.x + threadIdx.x;
    if (idx >= NN * HH) return;
    int c = idx % HH;
    float v = gamma[c] * (h[idx] - mu[c]) * rsqrtf(var[c] + BN_EPS) + beta[c];
    out[idx] = __float2bfloat16(fmaxf(v, 0.f));
}

// ---------------------------------------------------------------------------
// out = softmax([mlp_w, gcn_w])[0]*out_mlp + [1]*out_gcn
// ---------------------------------------------------------------------------
__global__ __launch_bounds__(256)
void combine_kernel(const float* __restrict__ out_mlp,
                    const float* __restrict__ out_gcn,
                    const float* __restrict__ mlp_w,
                    const float* __restrict__ gcn_w,
                    float* __restrict__ out)
{
    int idx = blockIdx.x * blockDim.x + threadIdx.x;
    if (idx >= NN * CC) return;
    float a = mlp_w[0], b = gcn_w[0];
    float m = fmaxf(a, b);
    float e0 = __expf(a - m), e1 = __expf(b - m);
    float inv = 1.f / (e0 + e1);
    out[idx] = e0 * inv * out_mlp[idx] + e1 * inv * out_gcn[idx];
}

// ---------------------------------------------------------------------------
// Host-side orchestration
// ---------------------------------------------------------------------------
extern "C" void kernel_launch(void* const* d_in, const int* in_sizes, int n_in,
                              void* d_out, int out_size, void* d_ws, size_t ws_size,
                              hipStream_t stream)
{
    const float* X_pca   = (const float*)d_in[0];
    const float* X_struct= (const float*)d_in[1];
    const float* Y       = (const float*)d_in[2];
    const float* adj     = (const float*)d_in[3];
    const float* alpha_p = (const float*)d_in[4];
    const float* Wp = (const float*)d_in[5];  const float* bp = (const float*)d_in[6];
    const float* Ws = (const float*)d_in[7];  const float* bs = (const float*)d_in[8];
    const float* Wl = (const float*)d_in[9];  const float* bl = (const float*)d_in[10];
    const float* Wm1= (const float*)d_in[11]; const float* bm1= (const float*)d_in[12];
    const float* Wm2= (const float*)d_in[13]; const float* bm2= (const float*)d_in[14];
    const float* Wg1= (const float*)d_in[15]; const float* bg1= (const float*)d_in[16];
    const float* gamma=(const float*)d_in[17];const float* beta=(const float*)d_in[18];
    const float* Wg2= (const float*)d_in[19]; const float* bg2= (const float*)d_in[20];
    const float* mlp_w=(const float*)d_in[21];const float* gcn_w=(const float*)d_in[22];
    float* out = (float*)d_out;

    char* w = (char*)d_ws;
    auto alloc = [&](size_t bytes) {
        void* p = (void*)w;
        w += (bytes + 255) & ~(size_t)255;
        return p;
    };
    __hip_bfloat16* Ahat   = (__hip_bfloat16*)alloc((size_t)NN * NN * 2);
    __hip_bfloat16* Agcn   = (__hip_bfloat16*)alloc((size_t)NN * NN * 2);
    float* dinv            = (float*)alloc(NN * 4);
    float* lblcomb         = (float*)alloc((size_t)NN * LBL * 4);
    __hip_bfloat16* BpS    = (__hip_bfloat16*)alloc((size_t)NN * CC * 2);   // K=8192,N=16
    __hip_bfloat16* combBf = (__hip_bfloat16*)alloc((size_t)NN * COMB * 2);
    __hip_bfloat16* Wg1pk  = (__hip_bfloat16*)alloc((size_t)COMB * HH * 2);
    float* Xc1             = (float*)alloc((size_t)NN * HH * 4);
    __hip_bfloat16* BpBig  = (__hip_bfloat16*)alloc((size_t)NN * HH * 2);
    float* h_raw           = (float*)alloc((size_t)NN * HH * 4);
    float* mu              = (float*)alloc(HH * 4);
    float* var             = (float*)alloc(HH * 4);
    __hip_bfloat16* hBf    = (__hip_bfloat16*)alloc((size_t)NN * HH * 2);
    __hip_bfloat16* Wg2pk  = (__hip_bfloat16*)alloc((size_t)HH * CC * 2);
    float* Xg2             = (float*)alloc((size_t)NN * CC * 4);
    __hip_bfloat16* BpS2   = (__hip_bfloat16*)alloc((size_t)NN * CC * 2);
    float* out_gcn         = (float*)alloc((size_t)NN * CC * 4);
    __hip_bfloat16* Wm1pk  = (__hip_bfloat16*)alloc((size_t)COMB * HH * 2);
    __hip_bfloat16* m1Bf   = (__hip_bfloat16*)alloc((size_t)NN * HH * 2);
    __hip_bfloat16* Wm2pk  = (__hip_bfloat16*)alloc((size_t)HH * CC * 2);
    float* out_mlp         = (float*)alloc((size_t)NN * CC * 4);

    const int Mt = NN / 16;   // 512 row tiles

    // 1) adjacency, degrees
    build_adj_kernel<<<NN, 256, 0, stream>>>(adj, Ahat, Agcn, dinv);

    // 2) 4-hop damped diffusion with row softmax
    for (int hop = 0; hop < 4; ++hop) {
        const float* src = (hop == 0) ? Y : lblcomb;
        int ld   = (hop == 0) ? CC : LBL;
        int col0 = (hop == 0) ? 0 : (hop - 1) * CC;
        int totalP = (CC / 16) * (NN / 32) * 256;
        pack_b_kernel<<<(totalP + 255) / 256, 256, 0, stream>>>(
            src, NN, CC, ld, col0, 0, nullptr, BpS);
        diffusion_kernel<<<Mt / 8, 256, 0, stream>>>(
            Ahat, BpS, Y, alpha_p, lblcomb, hop);
    }

    // 3) embeddings -> comb (bf16)
    comb_kernel<<<(NN * COMB + 255) / 256, 256, 0, stream>>>(
        X_pca, X_struct, lblcomb, Wp, bp, Ws, bs, Wl, bl, combBf);

    // 4) Xc1 = comb @ Wg1^T   (Ntiles=16, NT=4 -> 4 col-tiles per wave)
    {
        int totalP = (HH / 16) * (COMB / 32) * 256;
        pack_b_kernel<<<(totalP + 255) / 256, 256, 0, stream>>>(
            Wg1, COMB, HH, COMB, 0, 1, nullptr, Wg1pk);
        int waves = Mt * (HH / 16 / 4);
        gemm_bf16_kernel<4><<<waves / 8, 256, 0, stream>>>(
            combBf, COMB, Wg1pk, Mt, COMB, HH / 16,
            nullptr, nullptr, 0, Xc1, nullptr, HH);
    }

    // 5) h_raw = dinv .* (Agcn @ (dinv .* Xc1)) + bg1
    {
        int totalP = (HH / 16) * (NN / 32) * 256;
        pack_b_kernel<<<(totalP + 255) / 256, 256, 0, stream>>>(
            Xc1, NN, HH, HH, 0, 0, dinv, BpBig);
        int waves = Mt * (HH / 16 / 4);
        gemm_bf16_kernel<4><<<waves / 8, 256, 0, stream>>>(
            Agcn, NN, BpBig, Mt, NN, HH / 16,
            dinv, bg1, 0, h_raw, nullptr, HH);
    }

    // 6) batch norm + relu -> hBf
    bn_stats_kernel<<<HH, 256, 0, stream>>>(h_raw, mu, var);
    bn_apply_kernel<<<(NN * HH + 255) / 256, 256, 0, stream>>>(
        h_raw, mu, var, gamma, beta, hBf);

    // 7) Xg2 = h @ Wg2^T ; out_gcn = dinv .* (Agcn @ (dinv .* Xg2)) + bg2
    {
        int totalP = (CC / 16) * (HH / 32) * 256;
        pack_b_kernel<<<(totalP + 255) / 256, 256, 0, stream>>>(
            Wg2, HH, CC, HH, 0, 1, nullptr, Wg2pk);
        gemm_bf16_kernel<1><<<Mt / 8, 256, 0, stream>>>(
            hBf, HH, Wg2pk, Mt, HH, CC / 16,
            nullptr, nullptr, 0, Xg2, nullptr, CC);
        int totalP2 = (CC / 16) * (NN / 32) * 256;
        pack_b_kernel<<<(totalP2 + 255) / 256, 256, 0, stream>>>(
            Xg2, NN, CC, CC, 0, 0, dinv, BpS2);
        gemm_bf16_kernel<1><<<Mt / 8, 256, 0, stream>>>(
            Agcn, NN, BpS2, Mt, NN, CC / 16,
            dinv, bg2, 0, out_gcn, nullptr, CC);
    }

    // 8) MLP branch: m1 = relu(comb @ Wm1^T + bm1); out_mlp = m1 @ Wm2^T + bm2
    {
        int totalP = (HH / 16) * (COMB / 32) * 256;
        pack_b_kernel<<<(totalP + 255) / 256, 256, 0, stream>>>(
            Wm1, COMB, HH, COMB, 0, 1, nullptr, Wm1pk);
        int waves = Mt * (HH / 16 / 4);
        gemm_bf16_kernel<4><<<waves / 8, 256, 0, stream>>>(
            combBf, COMB, Wm1pk, Mt, COMB, HH / 16,
            nullptr, bm1, 1, nullptr, m1Bf, HH);
        int totalP2 = (CC / 16) * (HH / 32) * 256;
        pack_b_kernel<<<(totalP2 + 255) / 256, 256, 0, stream>>>(
            Wm2, HH, CC, HH, 0, 1, nullptr, Wm2pk);
        gemm_bf16_kernel<1><<<Mt / 8, 256, 0, stream>>>(
            m1Bf, HH, Wm2pk, Mt, HH, CC / 16,
            nullptr, bm2, 0, out_mlp, nullptr, CC);
    }

    // 9) weighted ensemble
    combine_kernel<<<(NN * CC + 255) / 256, 256, 0, stream>>>(
        out_mlp, out_gcn, mlp_w, gcn_w, out);
}
